// EdgeConv_81106162418022
// MI455X (gfx1250) — matrix operand order repaired and checked by reading, not compiled, not run on hardware
//
#include <hip/hip_runtime.h>
#include <hip/hip_bf16.h>
#include <hip/hip_fp16.h>

#define B_   4
#define C_   128
#define N_   2048
#define KNB  20      // neighbors == conv W dimension

typedef __attribute__((ext_vector_type(16))) _Float16 v16h;
typedef __attribute__((ext_vector_type(8)))  float    v8f;

// ---------------------------------------------------------------- fragment IO
__device__ __forceinline__ v16h ld2x16B(const _Float16* p0, const _Float16* p1) {
  union U { v16h v; float4 f[2]; } u;
  u.f[0] = *(const float4*)p0;
  u.f[1] = *(const float4*)p1;
  return u.v;
}

// A fragment (16x32 f16, MxK): lane<16 holds row M=lane, K {kk..kk+7, kk+16..kk+23};
// lane>=16 holds row M=lane-16, K {kk+8..kk+15, kk+24..kk+31}.
__device__ __forceinline__ v16h load_a_frag(const _Float16* base, int rowStride,
                                            int kk, int lane) {
  int half = lane >> 4, l = lane & 15;
  const _Float16* p = base + (size_t)l * rowStride + kk + half * 8;
  return ld2x16B(p, p + 16);
}

// B fragment (32x16 f16, KxN): lane holds column N=lane&15, K {kk..kk+7, kk+16..kk+23}
// (+8 for lanes 16..31). colbase = start of this lane's column data.
__device__ __forceinline__ v16h load_b_frag(const _Float16* colbase, int kk, int lane) {
  const _Float16* p = colbase + kk + (lane >> 4) * 8;
  return ld2x16B(p, p + 16);
}

// --------------------------------------------------- 0) x -> f16 (N,C) + ||x||^2
__global__ void prep_kernel(const float* __restrict__ x, _Float16* __restrict__ xh,
                            float* __restrict__ xx) {
  int bn = blockIdx.x;           // B*N blocks
  int b = bn / N_, n = bn % N_;
  int c = threadIdx.x;           // 128 threads
  float v = x[((size_t)b * C_ + c) * N_ + n];
  xh[((size_t)b * N_ + n) * C_ + c] = (_Float16)v;
  __shared__ float red[128];
  red[c] = v * v;
  __syncthreads();
  for (int s = 64; s > 0; s >>= 1) {
    if (c < s) red[c] += red[c + s];
    __syncthreads();
  }
  if (c == 0) xx[bn] = red[0];
}

// --------------------------------------------------- 1) pdist via WMMA f16
__global__ __launch_bounds__(256) void pdist_kernel(const _Float16* __restrict__ xh,
                                                    const float* __restrict__ xx,
                                                    float* __restrict__ pd) {
  const int TN = N_ / 16;                       // 128 tiles per dim
  int wid = blockIdx.x * (blockDim.x >> 5) + (threadIdx.x >> 5);
  int lane = threadIdx.x & 31;
  int b  = wid / (TN * TN);
  int r0 = wid % (TN * TN);
  int tn = r0 / TN, tm = r0 % TN;
  if (b >= B_) return;

  const _Float16* xb = xh + (size_t)b * N_ * C_;
  const _Float16* colbase = xb + (size_t)(tm * 16 + (lane & 15)) * C_;
  v8f acc = {};
  #pragma unroll
  for (int kk = 0; kk < C_; kk += 32) {
    v16h a  = load_a_frag(xb + (size_t)(tn * 16) * C_, C_, kk, lane);
    v16h bf = load_b_frag(colbase, kk, lane);
    acc = __builtin_amdgcn_wmma_f32_16x16x32_f16(false, a, false, bf,
                                                 (short)0, acc, false, false);
  }
  int col = tm * 16 + (lane & 15);
  float xxm = xx[b * N_ + col];
  float* pdb = pd + (size_t)b * N_ * N_;
  #pragma unroll
  for (int r = 0; r < 8; r++) {
    int row = tn * 16 + r + ((lane >> 4) << 3);
    pdb[(size_t)row * N_ + col] = 2.0f * acc[r] - xx[b * N_ + row] - xxm;
  }
}

// --------------------------------------------------- 2) top-k (20 argmax passes)
__global__ void topk_kernel(const float* __restrict__ pd, int* __restrict__ idx) {
  __shared__ float row[N_];
  __shared__ float rmax[256];
  __shared__ int   rid[256];
  int bn = blockIdx.x, t = threadIdx.x;
  const float* prow = pd + (size_t)bn * N_;
  for (int i = t; i < N_; i += 256) row[i] = prow[i];
  __syncthreads();
  for (int k = 0; k < KNB; k++) {
    float m = -3.0e38f; int mi = 0;
    for (int i = t; i < N_; i += 256) {
      float v = row[i];
      if (v > m) { m = v; mi = i; }
    }
    rmax[t] = m; rid[t] = mi;
    __syncthreads();
    for (int s = 128; s > 0; s >>= 1) {
      if (t < s && rmax[t + s] > rmax[t]) { rmax[t] = rmax[t + s]; rid[t] = rid[t + s]; }
      __syncthreads();
    }
    if (t == 0) { idx[bn * KNB + k] = rid[0]; row[rid[0]] = -3.0e38f; }
    __syncthreads();
  }
}

// --------------------------------------------------- 3) gather + attention + x0
__global__ void attn_kernel(const float* __restrict__ x, const int* __restrict__ idx,
                            const float* __restrict__ aw1, const float* __restrict__ ab1,
                            const float* __restrict__ aw2, const float* __restrict__ ab2,
                            _Float16* __restrict__ act0) {
  int bn = blockIdx.x;
  int b = bn / N_, n = bn % N_;
  int c = threadIdx.x;                 // 128 threads == channels
  __shared__ float diff[KNB][C_];
  __shared__ float h[KNB][64];
  __shared__ int sidx[KNB];
  if (c < KNB) sidx[c] = idx[bn * KNB + c];
  __syncthreads();

  float center = x[((size_t)b * C_ + c) * N_ + n];
  for (int k = 0; k < KNB; k++) {
    float f = x[((size_t)b * C_ + c) * N_ + sidx[k]];
    diff[k][c] = f - center;
  }
  __syncthreads();

  // h = relu(diff @ aw1 + ab1) : 20x64 outputs, 10 per thread
  for (int i = 0; i < 10; i++) {
    int o = c + 128 * i;
    int k = o >> 6, j = o & 63;
    float s = ab1[j];
    for (int cc = 0; cc < C_; cc++) s += diff[k][cc] * aw1[cc * 64 + j];
    h[k][j] = fmaxf(s, 0.0f);
  }
  __syncthreads();

  float att[KNB];
  float m = -3.0e38f;
  for (int k = 0; k < KNB; k++) {
    float s = ab2[c];
    for (int j = 0; j < 64; j++) s += h[k][j] * aw2[j * 128 + c];
    att[k] = s;
    m = fmaxf(m, s);
  }
  float sum = 0.0f;
  for (int k = 0; k < KNB; k++) { att[k] = __expf(att[k] - m); sum += att[k]; }
  float inv = 1.0f / sum;

  for (int k = 0; k < KNB; k++) {
    float feat = diff[k][c] + center;
    float wgt = feat * att[k] * inv;
    size_t base = ((size_t)bn * KNB + k) * 256;     // NHWC, Cin=256
    act0[base + c]       = (_Float16)(wgt - center);
    act0[base + 128 + c] = (_Float16)center;
  }
}

// --------------------------------------------------- 4) weight repack OIHW -> [Co][(tap)Ci] f16
__global__ void repack_kernel(const float* __restrict__ cw, _Float16* __restrict__ wk,
                              int CI, int CO) {
  int total = CO * CI * 9;
  for (int i = blockIdx.x * blockDim.x + threadIdx.x; i < total;
       i += gridDim.x * blockDim.x) {
    int co = i / (CI * 9);
    int r2 = i % (CI * 9);
    int tap = r2 / CI, ci = r2 % CI;
    wk[(size_t)co * (9 * CI) + tap * CI + ci] = (_Float16)cw[(size_t)(co * CI + ci) * 9 + tap];
  }
}

// --------------------------------------------------- 5) conv3x3 dilated, implicit GEMM WMMA
template <int CI, int CO, int DIL>
__global__ __launch_bounds__(256) void conv_kernel(const _Float16* __restrict__ act,
                                                   const _Float16* __restrict__ wk,
                                                   const float* __restrict__ cb,
                                                   float* __restrict__ y) {
  const int COT = CO / 16;
  const int PT = (N_ * KNB) / 16;      // 2560 pixel tiles per batch
  int wid = blockIdx.x * (blockDim.x >> 5) + (threadIdx.x >> 5);
  int lane = threadIdx.x & 31;
  int cot = wid % COT;
  int t = wid / COT;
  int pt = t % PT;
  int b = t / PT;
  if (b >= B_) return;

  int myp = pt * 16 + (lane & 15);     // this lane's output pixel (column)
  int h = myp / KNB, w = myp % KNB;
  const _Float16* arow = wk + (size_t)(cot * 16) * (9 * CI);

  v8f acc = {};
  #pragma unroll
  for (int tap = 0; tap < 9; tap++) {
    int hh = h + (tap / 3 - 1) * DIL;
    int ww = w + (tap % 3 - 1) * DIL;
    bool valid = ((unsigned)hh < (unsigned)N_) && ((unsigned)ww < (unsigned)KNB);
    const _Float16* colp =
        act + (((size_t)b * N_ + (valid ? hh : 0)) * KNB + (valid ? ww : 0)) * CI;
    if (valid) __builtin_prefetch((const void*)colp, 0, 0);
    #pragma unroll
    for (int cc = 0; cc < CI; cc += 32) {
      int kk = tap * CI + cc;
      v16h a = load_a_frag(arow, 9 * CI, kk, lane);
      v16h bf = {};
      if (valid) bf = load_b_frag(colp + cc, 0, lane);
      acc = __builtin_amdgcn_wmma_f32_16x16x32_f16(false, a, false, bf,
                                                   (short)0, acc, false, false);
    }
  }
  int co0 = cot * 16 + ((lane >> 4) << 3);
  float* yp = y + ((size_t)b * N_ * KNB + myp) * CO + co0;
  #pragma unroll
  for (int r = 0; r < 8; r++) yp[r] = acc[r] + cb[co0 + r];
}

// --------------------------------------------------- 6) BN stats (mean, rstd)
__global__ void stats_kernel(const float* __restrict__ y, float* __restrict__ stats,
                             int CO) {
  const int CNT = B_ * N_ * KNB;       // 163840
  int co = blockIdx.x, t = threadIdx.x;
  float s = 0.0f, sq = 0.0f;
  for (int i = t; i < CNT; i += 256) {
    float v = y[(size_t)i * CO + co];
    s += v; sq += v * v;
  }
  __shared__ float rs[256], rq[256];
  rs[t] = s; rq[t] = sq;
  __syncthreads();
  for (int k = 128; k > 0; k >>= 1) {
    if (t < k) { rs[t] += rs[t + k]; rq[t] += rq[t + k]; }
    __syncthreads();
  }
  if (t == 0) {
    float mean = rs[0] / (float)CNT;
    float var = rq[0] / (float)CNT - mean * mean;
    stats[co] = mean;
    stats[CO + co] = rsqrtf(var + 1e-5f);
  }
}

// --------------------------------------------------- 7) BN + ReLU -> f16 activation
__global__ void bnrelu_kernel(const float* __restrict__ y, const float* __restrict__ stats,
                              const float* __restrict__ g, const float* __restrict__ bb,
                              _Float16* __restrict__ act, int CO) {
  size_t total = (size_t)B_ * N_ * KNB * CO;
  for (size_t i = (size_t)blockIdx.x * blockDim.x + threadIdx.x; i < total;
       i += (size_t)gridDim.x * blockDim.x) {
    int c = (int)(i % CO);
    float v = (y[i] - stats[c]) * stats[CO + c] * g[c] + bb[c];
    act[i] = (_Float16)fmaxf(v, 0.0f);
  }
}

// --------------------------------------------------- 8) BN + ReLU + max over W
__global__ void final_kernel(const float* __restrict__ y, const float* __restrict__ stats,
                             const float* __restrict__ g, const float* __restrict__ bb,
                             float* __restrict__ out) {
  int bn = blockIdx.x;
  int b = bn / N_, n = bn % N_;
  int c = threadIdx.x;                 // 128 channels
  float mean = stats[c], rstd = stats[128 + c], gg = g[c], bv = bb[c];
  const float* yp = y + ((size_t)bn * KNB) * 128 + c;
  float m = 0.0f;
  for (int w = 0; w < KNB; w++) {
    float v = (yp[(size_t)w * 128] - mean) * rstd * gg + bv;
    m = fmaxf(m, fmaxf(v, 0.0f));
  }
  out[((size_t)b * 128 + c) * N_ + n] = m;
}

// ===================================================================== host
extern "C" void kernel_launch(void* const* d_in, const int* in_sizes, int n_in,
                              void* d_out, int out_size, void* d_ws, size_t ws_size,
                              hipStream_t stream) {
  (void)in_sizes; (void)n_in; (void)out_size; (void)ws_size;
  const float* x   = (const float*)d_in[0];
  const float* aw1 = (const float*)d_in[1];
  const float* ab1 = (const float*)d_in[2];
  const float* aw2 = (const float*)d_in[3];
  const float* ab2 = (const float*)d_in[4];

  char* ws = (char*)d_ws;
  size_t off = 0;
  auto alloc = [&](size_t bytes) -> char* {
    off = (off + 255) & ~(size_t)255;
    char* p = ws + off;
    off += bytes;
    return p;
  };

  _Float16* xh    = (_Float16*)alloc((size_t)B_ * N_ * C_ * 2);
  float*    xx    = (float*)   alloc((size_t)B_ * N_ * 4);
  float*    pd    = (float*)   alloc((size_t)B_ * N_ * N_ * 4);
  int*      idx   = (int*)     alloc((size_t)B_ * N_ * KNB * 4);
  _Float16* wk    = (_Float16*)alloc((size_t)160 * 1024);
  _Float16* act0  = (_Float16*)alloc((size_t)B_ * N_ * KNB * 256 * 2);
  _Float16* actA  = (_Float16*)alloc((size_t)B_ * N_ * KNB * 64 * 2);
  _Float16* actB  = (_Float16*)alloc((size_t)B_ * N_ * KNB * 64 * 2);
  float*    ybuf  = (float*)   alloc((size_t)B_ * N_ * KNB * 128 * 4);
  float*    stats = (float*)   alloc((size_t)2 * 128 * 4);
  float*    out   = (float*)d_out;

  // kNN graph
  prep_kernel<<<B_ * N_, 128, 0, stream>>>(x, xh, xx);
  pdist_kernel<<<(B_ * 128 * 128) / 8, 256, 0, stream>>>(xh, xx, pd);
  topk_kernel<<<B_ * N_, 256, 0, stream>>>(pd, idx);
  attn_kernel<<<B_ * N_, 128, 0, stream>>>(x, idx, aw1, ab1, aw2, ab2, act0);

  // conv stack: (ci, co, dil), in/out activation buffers
  const int PT = (N_ * KNB) / 16;

  // layer 1: 256 -> 32, d=1   (in: act0, out act: actA)
  repack_kernel<<<288, 256, 0, stream>>>((const float*)d_in[5], wk, 256, 32);
  conv_kernel<256, 32, 1><<<(B_ * PT * 2) / 8, 256, 0, stream>>>(
      act0, wk, (const float*)d_in[6], ybuf);
  stats_kernel<<<32, 256, 0, stream>>>(ybuf, stats, 32);
  bnrelu_kernel<<<4096, 256, 0, stream>>>(ybuf, stats, (const float*)d_in[7],
                                          (const float*)d_in[8], actA, 32);

  // layer 2: 32 -> 32, d=2    (actA -> actB)
  repack_kernel<<<36, 256, 0, stream>>>((const float*)d_in[9], wk, 32, 32);
  conv_kernel<32, 32, 2><<<(B_ * PT * 2) / 8, 256, 0, stream>>>(
      actA, wk, (const float*)d_in[10], ybuf);
  stats_kernel<<<32, 256, 0, stream>>>(ybuf, stats, 32);
  bnrelu_kernel<<<4096, 256, 0, stream>>>(ybuf, stats, (const float*)d_in[11],
                                          (const float*)d_in[12], actB, 32);

  // layer 3: 32 -> 64, d=5    (actB -> actA)
  repack_kernel<<<72, 256, 0, stream>>>((const float*)d_in[13], wk, 32, 64);
  conv_kernel<32, 64, 5><<<(B_ * PT * 4) / 8, 256, 0, stream>>>(
      actB, wk, (const float*)d_in[14], ybuf);
  stats_kernel<<<64, 256, 0, stream>>>(ybuf, stats, 64);
  bnrelu_kernel<<<4096, 256, 0, stream>>>(ybuf, stats, (const float*)d_in[15],
                                          (const float*)d_in[16], actA, 64);

  // layer 4: 64 -> 64, d=1    (actA -> actB)
  repack_kernel<<<144, 256, 0, stream>>>((const float*)d_in[17], wk, 64, 64);
  conv_kernel<64, 64, 1><<<(B_ * PT * 4) / 8, 256, 0, stream>>>(
      actA, wk, (const float*)d_in[18], ybuf);
  stats_kernel<<<64, 256, 0, stream>>>(ybuf, stats, 64);
  bnrelu_kernel<<<4096, 256, 0, stream>>>(ybuf, stats, (const float*)d_in[19],
                                          (const float*)d_in[20], actB, 64);

  // layer 5: 64 -> 64, d=2    (actB -> actA)
  repack_kernel<<<144, 256, 0, stream>>>((const float*)d_in[21], wk, 64, 64);
  conv_kernel<64, 64, 2><<<(B_ * PT * 4) / 8, 256, 0, stream>>>(
      actB, wk, (const float*)d_in[22], ybuf);
  stats_kernel<<<64, 256, 0, stream>>>(ybuf, stats, 64);
  bnrelu_kernel<<<4096, 256, 0, stream>>>(ybuf, stats, (const float*)d_in[23],
                                          (const float*)d_in[24], actA, 64);

  // layer 6: 64 -> 128, d=5   (actA -> ybuf), then BN+ReLU+max fused into output
  repack_kernel<<<288, 256, 0, stream>>>((const float*)d_in[25], wk, 64, 128);
  conv_kernel<64, 128, 5><<<(B_ * PT * 8) / 8, 256, 0, stream>>>(
      actA, wk, (const float*)d_in[26], ybuf);
  stats_kernel<<<128, 256, 0, stream>>>(ybuf, stats, 128);
  final_kernel<<<B_ * N_, 128, 0, stream>>>(ybuf, stats, (const float*)d_in[27],
                                            (const float*)d_in[28], out);
}
